// Dual_net_88639535055284
// MI455X (gfx1250) — matrix operand order repaired and checked by reading, not compiled, not run on hardware
//
#include <hip/hip_runtime.h>
#include <hip/hip_fp16.h>

#define HH 512
#define WWD 512
#define HWSZ (HH*WWD)          // 262144 = 2^18
#define NBATCH 2
#define CNT_PIX (NBATCH*HWSZ)  // 524288

typedef __attribute__((ext_vector_type(16))) _Float16 v16h;
typedef __attribute__((ext_vector_type(8)))  float    v8f;

// ---------------------------------------------------------------------------
// Kernel 1: feat = relu(conv3x3(img, w1)), aux16 = (f16)relu(conv3x3(img, wa))
// 3 -> 32 channels each, pad 1. Plane-major outputs.
// ---------------------------------------------------------------------------
__global__ __launch_bounds__(256)
void k_in_conv(const float* __restrict__ img,
               const float* __restrict__ w1,
               const float* __restrict__ wa,
               float* __restrict__ feat,
               _Float16* __restrict__ aux16)
{
  __shared__ float lw1[32*27];
  __shared__ float lwa[32*27];
  for (int i = threadIdx.x; i < 32*27; i += 256) { lw1[i] = w1[i]; lwa[i] = wa[i]; }
  __syncthreads();
  int p = blockIdx.x*256 + threadIdx.x;
  int n = p >> 18, y = (p >> 9) & 511, x = p & 511;
  float v[27];
  #pragma unroll
  for (int ci = 0; ci < 3; ++ci)
    #pragma unroll
    for (int dy = 0; dy < 3; ++dy)
      #pragma unroll
      for (int dx = 0; dx < 3; ++dx) {
        int yy = y + dy - 1, xx = x + dx - 1;
        float t = 0.f;
        if ((unsigned)yy < 512u && (unsigned)xx < 512u)
          t = img[((n*3+ci)<<18) + (yy<<9) + xx];
        v[(ci*3+dy)*3+dx] = t;
      }
  for (int co = 0; co < 32; ++co) {
    float a = 0.f, b = 0.f;
    const float* p1 = &lw1[co*27];
    const float* p2 = &lwa[co*27];
    #pragma unroll
    for (int k = 0; k < 27; ++k) { a = fmaf(v[k], p1[k], a); b = fmaf(v[k], p2[k], b); }
    a = fmaxf(a, 0.f); b = fmaxf(b, 0.f);
    feat [((n*32+co)<<18) + (y<<9) + x] = a;
    aux16[((n*32+co)<<18) + (y<<9) + x] = (_Float16)b;
  }
}

// ---------------------------------------------------------------------------
// Kernel 2: x1lin = conv3x3(feat, w_out), 32 -> 1, pad 1
// ---------------------------------------------------------------------------
__global__ __launch_bounds__(256)
void k_wout(const float* __restrict__ feat, const float* __restrict__ wout,
            float* __restrict__ x1lin)
{
  __shared__ float lw[288];
  for (int i = threadIdx.x; i < 288; i += 256) lw[i] = wout[i];
  __syncthreads();
  int p = blockIdx.x*256 + threadIdx.x;
  int n = p >> 18, y = (p >> 9) & 511, x = p & 511;
  float acc = 0.f;
  for (int ci = 0; ci < 32; ++ci) {
    const float* fp = &feat[((n*32+ci)<<18)];
    #pragma unroll
    for (int dy = 0; dy < 3; ++dy) {
      int yy = y + dy - 1; if ((unsigned)yy >= 512u) continue;
      #pragma unroll
      for (int dx = 0; dx < 3; ++dx) {
        int xx = x + dx - 1; if ((unsigned)xx >= 512u) continue;
        acc = fmaf(fp[(yy<<9)+xx], lw[ci*9 + dy*3 + dx], acc);
      }
    }
  }
  x1lin[p] = acc;
}

// ---------------------------------------------------------------------------
// Kernel 3: 32->32 3x3 conv as implicit GEMM with V_WMMA_F32_16X16X32_F16.
// Each block: 128 consecutive pixels (one n, one y, x-tile of 128), 8 waves.
// Wave w owns pixels [x0+16w, x0+16w+16) x all 32 output channels (2 N-tiles).
// K loop = 9 taps x 32 input channels. Also emits deterministic per-block
// partial sums for training-mode BatchNorm stats.
// Output zbuf is pixel-major [n][y][x][co].
// ---------------------------------------------------------------------------
__global__ __launch_bounds__(256)
void k_b1_wmma(const _Float16* __restrict__ aux16, const float* __restrict__ b1w,
               float* __restrict__ zbuf, float* __restrict__ psum,
               float* __restrict__ pssq)
{
  __shared__ __align__(16) _Float16 lin[3][130][32];   // rows y-1..y+1, x0-1..x0+128, ci
  __shared__ __align__(16) _Float16 lbw[9][32][32];    // [tap][co][ci]
  __shared__ float lpart[256][4];

  int blk = blockIdx.x;
  int n = blk >> 11;                 // 2048 blocks per batch
  int rem = blk & 2047;
  int y = rem >> 2;
  int x0 = (rem & 3) << 7;

  // stage weights: lbw[t][co][ci] = f16( b1w[(co*32+ci)*9 + t] )
  for (int i = threadIdx.x; i < 9*32*32; i += 256) {
    int t = i >> 10; int r = i & 1023; int co = r >> 5; int ci = r & 31;
    lbw[t][co][ci] = (_Float16)b1w[(co*32 + ci)*9 + t];
  }
  // stage input tile (zero-padded)
  for (int i = threadIdx.x; i < 32*3*130; i += 256) {
    int ci = i / (3*130); int r2 = i % (3*130); int row = r2 / 130; int xj = r2 % 130;
    int gy = y + row - 1, gx = x0 - 1 + xj;
    _Float16 val = (_Float16)0.0f;
    if ((unsigned)gy < 512u && (unsigned)gx < 512u)
      val = aux16[((n*32+ci)<<18) + (gy<<9) + gx];
    lin[row][xj][ci] = val;
  }
  __syncthreads();

  int lane = threadIdx.x & 31;
  int wv   = threadIdx.x >> 5;
  int m    = lane & 15;
  int kb   = (lane >> 4) << 4;   // A: K base 0 / 16
  int kbb  = (lane >> 4) << 4;   // B: K base 0 / 16

  v8f c0 = {}; v8f c1 = {};
  #pragma unroll
  for (int dy = 0; dy < 3; ++dy) {
    #pragma unroll
    for (int dx = 0; dx < 3; ++dx) {
      int t = dy*3 + dx;
      union { v16h h; uint4 u[2]; } A, B0, B1;
      int xi = wv*16 + m + dx;
      // A fragment: lane holds K kb..kb+15 ascending (matches B)
      A.u[0]  = *(const uint4*)&lin[dy][xi][kb];
      A.u[1]  = *(const uint4*)&lin[dy][xi][kb + 8];
      // B fragments: lane holds column co, K kbb..kbb+15 ascending
      B0.u[0] = *(const uint4*)&lbw[t][m][kbb];
      B0.u[1] = *(const uint4*)&lbw[t][m][kbb + 8];
      B1.u[0] = *(const uint4*)&lbw[t][16 + m][kbb];
      B1.u[1] = *(const uint4*)&lbw[t][16 + m][kbb + 8];
      c0 = __builtin_amdgcn_wmma_f32_16x16x32_f16(false, A.h, false, B0.h,
                                                  (short)0, c0, false, false);
      c1 = __builtin_amdgcn_wmma_f32_16x16x32_f16(false, A.h, false, B1.h,
                                                  (short)0, c1, false, false);
    }
  }

  // store C (pixel-major) + deterministic per-thread stat partials
  float s0 = 0.f, q0 = 0.f, s1 = 0.f, q1 = 0.f;
  int pixbase = (n<<18) + (y<<9) + x0 + wv*16;
  #pragma unroll
  for (int r = 0; r < 8; ++r) {
    int mm = r + ((lane >> 4) << 3);
    float v0 = c0[r], v1 = c1[r];
    zbuf[(pixbase + mm)*32 + m]      = v0;
    zbuf[(pixbase + mm)*32 + 16 + m] = v1;
    s0 += v0; q0 = fmaf(v0, v0, q0);
    s1 += v1; q1 = fmaf(v1, v1, q1);
  }
  lpart[threadIdx.x][0] = s0; lpart[threadIdx.x][1] = q0;
  lpart[threadIdx.x][2] = s1; lpart[threadIdx.x][3] = q1;
  __syncthreads();
  if (threadIdx.x < 32) {
    int ch = threadIdx.x;
    int slot = (ch < 16) ? 0 : 2;
    int l0 = ch & 15;
    float ss = 0.f, qq = 0.f;
    for (int w2 = 0; w2 < 8; ++w2) {
      int t0 = w2*32 + l0;
      ss += lpart[t0][slot]     + lpart[t0+16][slot];
      qq += lpart[t0][slot + 1] + lpart[t0+16][slot + 1];
    }
    psum[blockIdx.x*32 + ch] = ss;
    pssq[blockIdx.x*32 + ch] = qq;
  }
}

// ---------------------------------------------------------------------------
// Kernel 3b: finalize BN1 (biased var over N,H,W), produce per-channel a,b
// ---------------------------------------------------------------------------
__global__ void k_bn1_finalize(const float* __restrict__ psum,
                               const float* __restrict__ pssq,
                               const float* __restrict__ g,
                               const float* __restrict__ b,
                               float* __restrict__ bnA, float* __restrict__ bnB)
{
  int ch = threadIdx.x;
  if (ch >= 32) return;
  float s = 0.f, q = 0.f;
  for (int i = 0; i < 4096; ++i) { s += psum[i*32 + ch]; q += pssq[i*32 + ch]; }
  float cnt = (float)CNT_PIX;
  float mean = s / cnt;
  float var  = q / cnt - mean*mean;
  float inv  = rsqrtf(var + 1e-5f);
  float a = g[ch] * inv;
  bnA[ch] = a; bnB[ch] = b[ch] - mean*a;
}

// ---------------------------------------------------------------------------
// Kernel 4a: t = sum_c relu(bn1(z_c)) * b2w_c  (1x1 conv), + stats of t
// ---------------------------------------------------------------------------
__global__ __launch_bounds__(256)
void k_b2(const float* __restrict__ zbuf, const float* __restrict__ bnA,
          const float* __restrict__ bnB, const float* __restrict__ b2w,
          float* __restrict__ tbuf, float* __restrict__ tpart)
{
  __shared__ float la[32], lb[32], lw[32];
  __shared__ float red[256][2];
  if (threadIdx.x < 32) {
    la[threadIdx.x] = bnA[threadIdx.x];
    lb[threadIdx.x] = bnB[threadIdx.x];
    lw[threadIdx.x] = b2w[threadIdx.x];
  }
  __syncthreads();
  int p = blockIdx.x*256 + threadIdx.x;
  const float* zr = &zbuf[(size_t)p*32];
  float t = 0.f;
  #pragma unroll
  for (int c = 0; c < 32; ++c) {
    float v = fmaf(la[c], zr[c], lb[c]);
    v = fmaxf(v, 0.f);
    t = fmaf(v, lw[c], t);
  }
  tbuf[p] = t;
  red[threadIdx.x][0] = t; red[threadIdx.x][1] = t*t;
  __syncthreads();
  for (int s2 = 128; s2 > 0; s2 >>= 1) {
    if (threadIdx.x < s2) {
      red[threadIdx.x][0] += red[threadIdx.x + s2][0];
      red[threadIdx.x][1] += red[threadIdx.x + s2][1];
    }
    __syncthreads();
  }
  if (threadIdx.x == 0) {
    tpart[blockIdx.x*2]     = red[0][0];
    tpart[blockIdx.x*2 + 1] = red[0][1];
  }
}

__global__ void k_bn2_finalize(const float* __restrict__ tpart,
                               const float* __restrict__ g2,
                               const float* __restrict__ b2,
                               float* __restrict__ ab)
{
  if (threadIdx.x != 0) return;
  float s = 0.f, q = 0.f;
  for (int i = 0; i < 2048; ++i) { s += tpart[i*2]; q += tpart[i*2 + 1]; }
  float cnt = (float)CNT_PIX;
  float mean = s / cnt, var = q / cnt - mean*mean;
  float inv = rsqrtf(var + 1e-5f);
  float a = g2[0]*inv;
  ab[0] = a; ab[1] = b2[0] - mean*a;
}

__global__ __launch_bounds__(256)
void k_sum_s(const float* __restrict__ x1lin, const float* __restrict__ tbuf,
             const float* __restrict__ ab, float* __restrict__ sbuf)
{
  int p = blockIdx.x*256 + threadIdx.x;
  float v = fmaf(ab[0], tbuf[p], ab[1]);
  v = fmaxf(v, 0.f);
  sbuf[p] = x1lin[p] + v;
}

// ---------------------------------------------------------------------------
// Kernel 5: xm9[t] = s(p+off3(t))*s(p) + conv3x3(s, d0w)[t], 9 channels
// pixel-major [p][9]
// ---------------------------------------------------------------------------
__global__ __launch_bounds__(256)
void k_xm9(const float* __restrict__ sbuf, const float* __restrict__ d0w,
           float* __restrict__ xm9)
{
  __shared__ float lw[81];
  if (threadIdx.x < 81) lw[threadIdx.x] = d0w[threadIdx.x];
  __syncthreads();
  int p = blockIdx.x*256 + threadIdx.x;
  int n = p >> 18, y = (p >> 9) & 511, x = p & 511;
  float nb[9];
  #pragma unroll
  for (int dy = 0; dy < 3; ++dy)
    #pragma unroll
    for (int dx = 0; dx < 3; ++dx) {
      int yy = y + dy - 1, xx = x + dx - 1;
      nb[dy*3+dx] = ((unsigned)yy < 512u && (unsigned)xx < 512u)
                    ? sbuf[(n<<18) + (yy<<9) + xx] : 0.f;
    }
  float ctr = nb[4];
  float* op = &xm9[(size_t)p*9];
  #pragma unroll
  for (int t = 0; t < 9; ++t) {
    float acc = nb[t]*ctr;
    #pragma unroll
    for (int k = 0; k < 9; ++k) acc = fmaf(nb[k], lw[t*9 + k], acc);
    op[t] = acc;
  }
}

// ---------------------------------------------------------------------------
// Kernel 6: xm25[t] = s(p+off5(t))*s(p) + conv3x3(xm9, d1w)[t], 25 channels
// ---------------------------------------------------------------------------
__global__ __launch_bounds__(256)
void k_xm25(const float* __restrict__ sbuf, const float* __restrict__ xm9,
            const float* __restrict__ d1w, float* __restrict__ xm25)
{
  __shared__ float lw[2025];
  for (int i = threadIdx.x; i < 2025; i += 256) lw[i] = d1w[i];
  __syncthreads();
  int p = blockIdx.x*256 + threadIdx.x;
  int n = p >> 18, y = (p >> 9) & 511, x = p & 511;
  float s5[25];
  #pragma unroll
  for (int dy = 0; dy < 5; ++dy)
    #pragma unroll
    for (int dx = 0; dx < 5; ++dx) {
      int yy = y + dy - 2, xx = x + dx - 2;
      s5[dy*5+dx] = ((unsigned)yy < 512u && (unsigned)xx < 512u)
                    ? sbuf[(n<<18) + (yy<<9) + xx] : 0.f;
    }
  float ctr = s5[12];
  float nb[9][9];
  #pragma unroll
  for (int u = 0; u < 3; ++u)
    #pragma unroll
    for (int v = 0; v < 3; ++v) {
      int yy = y + u - 1, xx = x + v - 1;
      bool ok = ((unsigned)yy < 512u && (unsigned)xx < 512u);
      const float* q = ok ? &xm9[(size_t)((n<<18) + (yy<<9) + xx)*9] : (const float*)0;
      #pragma unroll
      for (int c = 0; c < 9; ++c) nb[u*3+v][c] = ok ? q[c] : 0.f;
    }
  float* op = &xm25[(size_t)p*25];
  for (int t = 0; t < 25; ++t) {
    float acc = s5[t]*ctr;
    const float* wt = &lw[t*81];
    #pragma unroll
    for (int ci = 0; ci < 9; ++ci)
      #pragma unroll
      for (int k = 0; k < 9; ++k)
        acc = fmaf(nb[k][ci], wt[ci*9 + k], acc);
    op[t] = acc;
  }
}

// ---------------------------------------------------------------------------
// Kernel 7: Conv4D over (5x5 channel grid)x(3x3 spatial) + affinity-weighted
// sum -> seg = [1-out, out]
// ---------------------------------------------------------------------------
__global__ __launch_bounds__(256)
void k_conv4d(const float* __restrict__ sbuf, const float* __restrict__ xm25,
              const float* __restrict__ d4w, float* __restrict__ outp)
{
  __shared__ float lw[81];
  if (threadIdx.x < 81) lw[threadIdx.x] = d4w[threadIdx.x];
  __syncthreads();
  int p = blockIdx.x*256 + threadIdx.x;
  int n = p >> 18, y = (p >> 9) & 511, x = p & 511;
  float acc[25];
  #pragma unroll
  for (int t = 0; t < 25; ++t) acc[t] = 0.f;
  for (int kh = 0; kh < 3; ++kh)
    for (int kw = 0; kw < 3; ++kw) {
      int yy = y + kh - 1, xx = x + kw - 1;
      bool ok = ((unsigned)yy < 512u && (unsigned)xx < 512u);
      const float* q = ok ? &xm25[(size_t)((n<<18) + (yy<<9) + xx)*25] : (const float*)0;
      float v[25];
      #pragma unroll
      for (int c = 0; c < 25; ++c) v[c] = ok ? q[c] : 0.f;
      int kk = kh*3 + kw;
      #pragma unroll
      for (int d = 0; d < 5; ++d)
        #pragma unroll
        for (int e = 0; e < 5; ++e) {
          float a = 0.f;
          #pragma unroll
          for (int kd = 0; kd < 3; ++kd) {
            int dd = d + kd - 1; if ((unsigned)dd >= 5u) continue;
            #pragma unroll
            for (int ke = 0; ke < 3; ++ke) {
              int ee = e + ke - 1; if ((unsigned)ee >= 5u) continue;
              a = fmaf(v[dd*5 + ee], lw[(kd*3 + ke)*9 + kk], a);
            }
          }
          acc[d*5 + e] += a;
        }
    }
  float out = 0.f;
  #pragma unroll
  for (int d = 0; d < 5; ++d)
    #pragma unroll
    for (int e = 0; e < 5; ++e) {
      int yy = y + d - 2, xx = x + e - 2;
      float sv = ((unsigned)yy < 512u && (unsigned)xx < 512u)
                 ? sbuf[(n<<18) + (yy<<9) + xx] : 0.f;
      out = fmaf(sv, acc[d*5 + e], out);
    }
  out *= (1.f/25.f);
  outp[((size_t)(n*2)   << 18) + (y<<9) + x] = 1.f - out;
  outp[((size_t)(n*2+1) << 18) + (y<<9) + x] = out;
}

// ---------------------------------------------------------------------------
// results_erasing: exact 2-pass radix select of the k-th largest per (n,c)
// ---------------------------------------------------------------------------
__device__ __forceinline__ unsigned fkey(float f) {
  unsigned u = __float_as_uint(f);
  return (u & 0x80000000u) ? ~u : (u | 0x80000000u);
}
__device__ __forceinline__ float funkey(unsigned u) {
  unsigned r = (u & 0x80000000u) ? (u ^ 0x80000000u) : ~u;
  return __uint_as_float(r);
}

__global__ __launch_bounds__(256)
void k_hist_hi(const float* __restrict__ seg, unsigned* __restrict__ hist)
{
  int i = blockIdx.x*256 + threadIdx.x;  // 4*HWSZ elements
  int nc = i >> 18;
  unsigned u = fkey(seg[i]);
  atomicAdd(&hist[(nc << 16) + (u >> 16)], 1u);
}

__global__ void k_sel_hi(const unsigned* __restrict__ hist,
                         const float* __restrict__ ratio, int* __restrict__ sel)
{
  int nc = threadIdx.x;
  if (nc >= 4) return;
  int n = nc >> 1;
  float fp = floorf(ratio[n] * (float)HWSZ);
  int kk = (int)floorf(fp * 0.1f);
  int rank = kk < 1 ? 1 : kk;           // idx = max(k-1,0) -> rank = max(k,1)
  const unsigned* h = &hist[nc << 16];
  int cum = 0, bin = 0, remv = rank;
  for (int b2 = 65535; b2 >= 0; --b2) {
    int c = (int)h[b2];
    if (cum + c >= rank) { bin = b2; remv = rank - cum; break; }
    cum += c;
  }
  sel[nc*2] = bin; sel[nc*2 + 1] = remv;
}

__global__ __launch_bounds__(256)
void k_hist_lo(const float* __restrict__ seg, const int* __restrict__ sel,
               unsigned* __restrict__ hist)
{
  int i = blockIdx.x*256 + threadIdx.x;
  int nc = i >> 18;
  unsigned u = fkey(seg[i]);
  if ((int)(u >> 16) == sel[nc*2])
    atomicAdd(&hist[(nc << 16) + (u & 0xFFFFu)], 1u);
}

__global__ void k_sel_lo(const unsigned* __restrict__ hist,
                         const int* __restrict__ sel, float* __restrict__ thr)
{
  int nc = threadIdx.x;
  if (nc >= 4) return;
  const unsigned* h = &hist[nc << 16];
  int remv = sel[nc*2 + 1];
  int cum = 0; unsigned lo = 0;
  for (int b2 = 65535; b2 >= 0; --b2) {
    int c = (int)h[b2];
    if (cum + c >= remv) { lo = (unsigned)b2; break; }
    cum += c;
  }
  unsigned u = ((unsigned)sel[nc*2] << 16) | lo;
  thr[nc] = funkey(u);
}

__global__ __launch_bounds__(256)
void k_erase(const float* __restrict__ seg, const float* __restrict__ thr,
             const float* __restrict__ img, float* __restrict__ out)
{
  int p = blockIdx.x*256 + threadIdx.x;  // NBATCH*HWSZ
  int n = p >> 18; int off = p & (HWSZ - 1);
  bool m = (seg[((size_t)(n*2)   << 18) + off] > thr[n*2]) ||
           (seg[((size_t)(n*2+1) << 18) + off] > thr[n*2 + 1]);
  #pragma unroll
  for (int c = 0; c < 3; ++c) {
    float v = img[((n*3 + c) << 18) + off];
    out[((n*3 + c) << 18) + off] = m ? 0.f : v;
  }
}

// ---------------------------------------------------------------------------
// Host orchestration
// ---------------------------------------------------------------------------
static inline size_t alignup256(size_t v) { return (v + 255) & ~(size_t)255; }

extern "C" void kernel_launch(void* const* d_in, const int* in_sizes, int n_in,
                              void* d_out, int out_size, void* d_ws, size_t ws_size,
                              hipStream_t stream)
{
  const float* x       = (const float*)d_in[0];
  const float* ratio   = (const float*)d_in[1];
  const float* unet_w1 = (const float*)d_in[2];
  const float* unet_wa = (const float*)d_in[3];
  const float* w_out   = (const float*)d_in[4];
  const float* b1_w    = (const float*)d_in[5];
  const float* b1_g    = (const float*)d_in[6];
  const float* b1_b    = (const float*)d_in[7];
  const float* b2_w    = (const float*)d_in[8];
  const float* b2_g    = (const float*)d_in[9];
  const float* b2_b    = (const float*)d_in[10];
  const float* d0_w    = (const float*)d_in[11];
  const float* d1_w    = (const float*)d_in[12];
  const float* d4_w    = (const float*)d_in[13];

  char* ws = (char*)d_ws;
  size_t off = 0;
  auto alloc = [&](size_t bytes) -> char* {
    char* p = ws + off; off = alignup256(off + bytes); return p;
  };

  // Region A: feat (67MB) later reused as xm25 (52MB)
  float*    feat  = (float*)alloc((size_t)NBATCH*32*HWSZ*4);
  float*    xm25  = feat;
  // Region B: aux16 f16
  _Float16* aux16 = (_Float16*)alloc((size_t)NBATCH*32*HWSZ*2);
  // Region C: zbuf (67MB) later reused as xm9 (19MB)
  float*    zbuf  = (float*)alloc((size_t)NBATCH*HWSZ*32*4);
  float*    xm9   = zbuf;
  float*    x1lin = (float*)alloc((size_t)NBATCH*HWSZ*4);
  float*    tbuf  = (float*)alloc((size_t)NBATCH*HWSZ*4);
  float*    sbuf  = (float*)alloc((size_t)NBATCH*HWSZ*4);
  float*    x2img = (float*)alloc((size_t)NBATCH*3*HWSZ*4);
  float*    psum  = (float*)alloc((size_t)4096*32*4);
  float*    pssq  = (float*)alloc((size_t)4096*32*4);
  float*    bnA   = (float*)alloc(32*4);
  float*    bnB   = (float*)alloc(32*4);
  float*    tpart = (float*)alloc((size_t)2048*2*4);
  float*    ab2   = (float*)alloc(2*4);
  float*    thr   = (float*)alloc(4*4);
  int*      sel   = (int*)alloc(8*4);
  unsigned* hist  = (unsigned*)alloc((size_t)4*65536*4);

  auto run_pipeline = [&](const float* img, float* outp) {
    k_in_conv    <<<2048, 256, 0, stream>>>(img, unet_w1, unet_wa, feat, aux16);
    k_wout       <<<2048, 256, 0, stream>>>(feat, w_out, x1lin);
    k_b1_wmma    <<<4096, 256, 0, stream>>>(aux16, b1_w, zbuf, psum, pssq);
    k_bn1_finalize<<<1, 32, 0, stream>>>(psum, pssq, b1_g, b1_b, bnA, bnB);
    k_b2         <<<2048, 256, 0, stream>>>(zbuf, bnA, bnB, b2_w, tbuf, tpart);
    k_bn2_finalize<<<1, 1, 0, stream>>>(tpart, b2_g, b2_b, ab2);
    k_sum_s      <<<2048, 256, 0, stream>>>(x1lin, tbuf, ab2, sbuf);
    k_xm9        <<<2048, 256, 0, stream>>>(sbuf, d0_w, xm9);
    k_xm25       <<<2048, 256, 0, stream>>>(sbuf, xm9, d1_w, xm25);
    k_conv4d     <<<2048, 256, 0, stream>>>(sbuf, xm25, d4_w, outp);
  };

  float* out1 = (float*)d_out;
  float* out2 = out1 + (size_t)NBATCH*2*HWSZ;

  run_pipeline(x, out1);

  hipMemsetAsync(hist, 0, (size_t)4*65536*4, stream);
  k_hist_hi<<<4096, 256, 0, stream>>>(out1, hist);
  k_sel_hi <<<1, 32, 0, stream>>>(hist, ratio, sel);
  hipMemsetAsync(hist, 0, (size_t)4*65536*4, stream);
  k_hist_lo<<<4096, 256, 0, stream>>>(out1, sel, hist);
  k_sel_lo <<<1, 32, 0, stream>>>(hist, sel, thr);
  k_erase  <<<2048, 256, 0, stream>>>(out1, thr, x, x2img);

  run_pipeline(x2img, out2);
}
